// MultiheadAttention_60327110640352
// MI455X (gfx1250) — compile-verified, matrix operand order checked by reading, not run on hardware
//
#include <hip/hip_runtime.h>
#include <hip/hip_bf16.h>
#include <math.h>
#include <stdint.h>

// ---------------------------------------------------------------------------
// Types for gfx1250 WMMA (wave32)
// ---------------------------------------------------------------------------
typedef __bf16 bf16_t;
typedef __attribute__((ext_vector_type(16))) __bf16 v16bf;
typedef __attribute__((ext_vector_type(8)))  __bf16 v8bf;
typedef __attribute__((ext_vector_type(8)))  float  v8f;
typedef __attribute__((ext_vector_type(4)))  unsigned int v4u;
typedef __attribute__((ext_vector_type(8)))  int v8i;
typedef __attribute__((ext_vector_type(4)))  int v4i;

union Frag16 { v16bf v; v8bf h[2]; };

static __device__ __forceinline__ v8f wmma_bf16(v16bf a, v16bf b, v8f c) {
  return __builtin_amdgcn_wmma_f32_16x16x32_bf16(false, a, false, b,
                                                 (short)0, c, false, false);
}

// 16x32 bf16 fragment load (A layout; identical gather when src is [N][K]).
// elements 0..7  -> K = k0 + half*8 + e
// elements 8..15 -> K = k0 + 16 + half*8 + (e-8)
static __device__ __forceinline__ v16bf load_frag_b16(const bf16_t* base,
                                                      int row_stride, int k0) {
  const int lane = threadIdx.x & 31;
  const int r    = lane & 15;
  const int half = lane >> 4;
  const bf16_t* p = base + (size_t)r * row_stride + k0 + half * 8;
  Frag16 f;
  f.h[0] = *(const v8bf*)(p);
  f.h[1] = *(const v8bf*)(p + 16);
  return f.v;
}

// ---------------------------------------------------------------------------
// TDM: 2D tile load Global -> LDS (bf16 elements, data_size=2B).
// D# packing per CDNA5 ISA.
// ---------------------------------------------------------------------------
static __device__ __forceinline__ void tdm_load_2d_bf16(
    const void* gaddr, unsigned lds_off,
    unsigned tensor_dim0, unsigned tensor_dim1,
    unsigned tile_dim0, unsigned tile_dim1,
    unsigned long long dim0_stride,
    bool pad_en, unsigned pad_interval, unsigned pad_amount) {
  unsigned long long ga = (unsigned long long)(uintptr_t)gaddr;
  v4u g0;
  g0[0] = 1u;                                        // count=1, user mode
  g0[1] = lds_off;                                   // lds_addr
  g0[2] = (unsigned)(ga & 0xFFFFFFFFu);              // global_addr[31:0]
  g0[3] = (unsigned)((ga >> 32) & 0x01FFFFFFu) | (2u << 30);  // addr[56:32]|type=2
  v8i g1;
  unsigned w0 = (1u << 16);                          // data_size=1 -> 2 bytes
  if (pad_en) w0 |= (1u << 20) | (pad_interval << 22) | (pad_amount << 25);
  g1[0] = (int)w0;                                   // wg_mask=0, flags
  g1[1] = (int)((tensor_dim0 & 0xFFFFu) << 16);      // dim0[15:0] @bits63:48
  g1[2] = (int)(((tensor_dim0 >> 16) & 0xFFFFu) |
                ((tensor_dim1 & 0xFFFFu) << 16));    // dim0[31:16], dim1[15:0]
  g1[3] = (int)(((tensor_dim1 >> 16) & 0xFFFFu) |
                ((tile_dim0 & 0xFFFFu) << 16));      // dim1[31:16], tile0
  g1[4] = (int)(tile_dim1 & 0xFFFFu);                // tile1; tile2=0
  g1[5] = (int)(unsigned)(dim0_stride & 0xFFFFFFFFu);
  g1[6] = (int)(unsigned)((dim0_stride >> 32) & 0xFFFFu);  // stride0 hi; stride1=0
  g1[7] = 0;
  v4i z4 = {0, 0, 0, 0};
#if defined(__clang_major__) && (__clang_major__ >= 23)
  v8i z8 = {0, 0, 0, 0, 0, 0, 0, 0};
  __builtin_amdgcn_tensor_load_to_lds(g0, g1, z4, z4, z8, 0);
#else
  __builtin_amdgcn_tensor_load_to_lds(g0, g1, z4, z4, 0);
#endif
}

// ---------------------------------------------------------------------------
// fp32 -> bf16 conversion
// ---------------------------------------------------------------------------
__global__ __launch_bounds__(256) void cvt_f32_bf16(const float* __restrict__ in,
                                                    bf16_t* __restrict__ out,
                                                    size_t n) {
  size_t i = (size_t)blockIdx.x * blockDim.x + threadIdx.x;
  if (i < n) out[i] = (bf16_t)in[i];
}

// ---------------------------------------------------------------------------
// 128x128 GEMM mainloop. A tile staged by the Tensor Data Mover (async DMA,
// ping-pong LDS buffers); B (weight) tile staged via registers with an
// in-register transpose to [n][k]. Block = 256 threads = 8 waves; wave
// (wm = w>>2, wn = w&3) owns 64x32 of C.
// ---------------------------------------------------------------------------
#define LDS_STRIDE 40
#define TILE_ELEMS (128 * LDS_STRIDE)

static __device__ __forceinline__ void gemm_mainloop(
    const bf16_t* __restrict__ A, const bf16_t* __restrict__ Bw,
    int lda, int ldb, int Ksz, int rows, int rowBase, int colBase,
    bf16_t* Xs, bf16_t* Wt, v8f acc[4][2]) {
  const int wave = threadIdx.x >> 5;
  const int wm = wave >> 2;   // 0..1
  const int wn = wave & 3;    // 0..3
  const int tid = threadIdx.x;

  int bkk[2], bc[2];
  #pragma unroll
  for (int it = 0; it < 2; ++it) {
    int i = tid + it * 256;
    bkk[it] = i >> 4;  bc[it] = i & 15;   // B: 32 k x 16 chunks of 8 cols
  }

  v8bf bReg[2];
  auto fetchB = [&](int k0) {
    #pragma unroll
    for (int it = 0; it < 2; ++it)
      bReg[it] = *(const v8bf*)(Bw + (size_t)(k0 + bkk[it]) * ldb + colBase + bc[it] * 8);
  };
  auto stageB = [&](int buf) {
    bf16_t* wt = Wt + buf * TILE_ELEMS;
    #pragma unroll
    for (int it = 0; it < 2; ++it)
      #pragma unroll
      for (int e = 0; e < 8; ++e)
        wt[(bc[it] * 8 + e) * LDS_STRIDE + bkk[it]] = bReg[it][e];   // transpose
  };
  // A tile 128x32 via TDM: row = 32 bf16 = 64B = 16 DW -> pad_interval code 3,
  // pad 16B = 4 DW -> pad_amount code 3 => LDS row stride 40 elements.
  auto tdmA = [&](int k0, int buf) {
    tdm_load_2d_bf16(A + (size_t)rowBase * lda + k0,
                     (unsigned)(uintptr_t)(Xs + buf * TILE_ELEMS),
                     /*dim0=*/(unsigned)lda, /*dim1=*/(unsigned)rows,
                     /*tile0=*/32u, /*tile1=*/128u,
                     /*stride0=*/(unsigned long long)lda,
                     /*pad=*/true, /*interval=*/3u, /*amount=*/3u);
  };

  if (wave == 0) tdmA(0, 0);
  fetchB(0);
  stageB(0);
  if (wave == 0) __builtin_amdgcn_s_wait_tensorcnt((short)0);
  __syncthreads();

  const int nTiles = Ksz >> 5;
  for (int kt = 0; kt < nTiles; ++kt) {
    const int buf = kt & 1;
    if (kt + 1 < nTiles) {
      if (wave == 0) tdmA((kt + 1) << 5, 1 - buf);   // async DMA during compute
      fetchB((kt + 1) << 5);                         // no wait yet
      if (kt + 2 < nTiles)
        __builtin_prefetch(Bw + (size_t)(((kt + 2) << 5) + bkk[0]) * ldb + colBase, 0, 1);
    }

    const bf16_t* xs = Xs + buf * TILE_ELEMS;
    const bf16_t* wt = Wt + buf * TILE_ELEMS;
    v16bf af[4], bfv[2];
    #pragma unroll
    for (int mt = 0; mt < 4; ++mt)
      af[mt] = load_frag_b16(&xs[(wm * 64 + mt * 16) * LDS_STRIDE], LDS_STRIDE, 0);
    #pragma unroll
    for (int nt = 0; nt < 2; ++nt)
      bfv[nt] = load_frag_b16(&wt[(wn * 32 + nt * 16) * LDS_STRIDE], LDS_STRIDE, 0);
    #pragma unroll
    for (int mt = 0; mt < 4; ++mt)
      #pragma unroll
      for (int nt = 0; nt < 2; ++nt)
        acc[mt][nt] = wmma_bf16(af[mt], bfv[nt], acc[mt][nt]);

    if (kt + 1 < nTiles) {
      stageB(1 - buf);                               // other buffer: safe
      if (wave == 0) __builtin_amdgcn_s_wait_tensorcnt((short)0);
      __syncthreads();
    }
  }
}

// ---------------------------------------------------------------------------
// Kernel 1: QKV projection, scatter to Q/K/V [B,H,S,Dh] bf16.
// ---------------------------------------------------------------------------
__global__ __launch_bounds__(256) void gemm_qkv_kernel(
    const bf16_t* __restrict__ X, const bf16_t* __restrict__ W,
    const float* __restrict__ bias,
    bf16_t* __restrict__ Q, bf16_t* __restrict__ K, bf16_t* __restrict__ V,
    int S, int Ksz, int ldb) {
  __shared__ bf16_t Xs[2 * TILE_ELEMS];
  __shared__ bf16_t Wt[2 * TILE_ELEMS];
  v8f acc[4][2] = {};
  const int rowBase = blockIdx.y * 128;
  const int colBase = blockIdx.x * 128;
  const int rows = (int)gridDim.y * 128;
  gemm_mainloop(X, W, Ksz, ldb, Ksz, rows, rowBase, colBase, Xs, Wt, acc);

  const int wave = threadIdx.x >> 5, lane = threadIdx.x & 31;
  const int wm = wave >> 2, wn = wave & 3;
  const int half = lane >> 4, n0 = lane & 15;
  #pragma unroll
  for (int mt = 0; mt < 4; ++mt) {
    #pragma unroll
    for (int nt = 0; nt < 2; ++nt) {
      int col = colBase + wn * 32 + nt * 16 + n0;
      float bv = bias[col];
      int sel = col >> 10;          // 0=Q 1=K 2=V
      int nn = col & 1023;
      int h = nn >> 6, d = nn & 63;
      bf16_t* dst = (sel == 0) ? Q : ((sel == 1) ? K : V);
      #pragma unroll
      for (int r = 0; r < 8; ++r) {
        int row = rowBase + wm * 64 + mt * 16 + r + 8 * half;
        int b = row / S, s = row % S;
        float v = acc[mt][nt][r] + bv;
        dst[(((size_t)(b * 16 + h)) * S + s) * 64 + d] = (bf16_t)v;
      }
    }
  }
}

// ---------------------------------------------------------------------------
// Kernel 2: RoPE on Q and K, layout [B*H, S, 64] bf16, pairs (2p, 2p+1).
// ---------------------------------------------------------------------------
__global__ __launch_bounds__(256) void rope_kernel(bf16_t* __restrict__ Q,
                                                   bf16_t* __restrict__ K,
                                                   int BH, int S) {
  int idx = blockIdx.x * blockDim.x + threadIdx.x;
  int total = BH * S * 32;
  if (idx >= 2 * total) return;
  bf16_t* T = (idx < total) ? Q : K;
  int i = (idx < total) ? idx : idx - total;
  int p  = i & 31;
  int s  = (i >> 5) % S;
  int bh = i / (32 * S);
  float inv = __powf(10000.0f, -((float)(2 * p)) / 64.0f);
  float ang = (float)s * inv;
  float sn, cs;
  __sincosf(ang, &sn, &cs);
  size_t base = ((size_t)bh * S + s) * 64 + 2 * p;
  float x0 = (float)T[base], x1 = (float)T[base + 1];
  T[base]     = (bf16_t)(x0 * cs - x1 * sn);
  T[base + 1] = (bf16_t)(x0 * sn + x1 * cs);
}

// ---------------------------------------------------------------------------
// Kernel 3: flash attention. 128 query rows/block of one (b,h); 8 waves x 16
// rows. K tile loaded by TDM (tensor_load_to_lds, padded rows), V transposed
// manually. Output bf16 to [B, S, H*64].
// ---------------------------------------------------------------------------
__global__ __launch_bounds__(256) void flash_attn_kernel(
    const bf16_t* __restrict__ Q, const bf16_t* __restrict__ K,
    const bf16_t* __restrict__ V, const unsigned char* __restrict__ padmask,
    const int* __restrict__ is_causal_p, bf16_t* __restrict__ attn,
    int S) {
  const int H = 16;
  const int bh = blockIdx.y;
  const int b = bh / H, h = bh % H;
  const int qb = blockIdx.x;
  const int wave = threadIdx.x >> 5, lane = threadIdx.x & 31;
  const int half = lane >> 4, n0 = lane & 15;

  __shared__ bf16_t Ks[64 * 72];       // [key][dh], padded rows (TDM target)
  __shared__ bf16_t Vt[64 * 72];       // [dh][key]
  __shared__ bf16_t Ps[8 * 16 * 72];   // per-wave P scratch [row][key]

  const bf16_t* Qb = Q + (size_t)bh * S * 64;
  const bf16_t* Kb = K + (size_t)bh * S * 64;
  const bf16_t* Vb = V + (size_t)bh * S * 64;
  const int row_base = qb * 128 + wave * 16;
  const bool causal = (is_causal_p[0] != 0);
  const unsigned ks_lds = (unsigned)(uintptr_t)(&Ks[0]);   // LDS aperture: low 32b

  v16bf qf[2];
  qf[0] = load_frag_b16(Qb + (size_t)row_base * 64, 64, 0);
  qf[1] = load_frag_b16(Qb + (size_t)row_base * 64, 64, 32);

  v8f oacc[4] = {};
  float m_i[8], l_i[8];
  #pragma unroll
  for (int r = 0; r < 8; ++r) { m_i[r] = -INFINITY; l_i[r] = 0.0f; }

  const int kv_end = causal ? (qb * 128 + 128 < S ? qb * 128 + 128 : S) : S;

  for (int kv = 0; kv < kv_end; kv += 64) {
    __syncthreads();   // previous iteration's LDS reads complete

    // K tile 64x64 via TDM into padded LDS rows:
    // row = 64 bf16 = 128B = 32 DW -> pad_interval code 4, pad 4 DW -> code 3.
    if (wave == 0) {
      tdm_load_2d_bf16(Kb + (size_t)kv * 64, ks_lds,
                       /*dim0=*/64, /*dim1=*/(unsigned)S,
                       /*tile0=*/64, /*tile1=*/64,
                       /*stride0=*/64ull,
                       /*pad=*/true, /*interval=*/4u, /*amount=*/3u);
    }

    // V tile: transposed -> Vt[dh][key] (cooperative, all waves)
    #pragma unroll
    for (int it = 0; it < 2; ++it) {
      int i = threadIdx.x + it * 256;
      int row = i >> 3, c = i & 7;
      v8bf val = *(const v8bf*)(Vb + (size_t)(kv + row) * 64 + c * 8);
      #pragma unroll
      for (int e = 0; e < 8; ++e) Vt[(c * 8 + e) * 72 + row] = val[e];
    }

    if (wave == 0) __builtin_amdgcn_s_wait_tensorcnt((short)0);
    __syncthreads();

    if (causal && kv > row_base + 15) continue;  // fully-masked tile (wave-uniform)

    // S = Q K^T  (4 key tiles of 16)
    v8f sc[4];
    #pragma unroll
    for (int kt = 0; kt < 4; ++kt) {
      v8f a = {};
      #pragma unroll
      for (int ks = 0; ks < 2; ++ks) {
        v16bf bfr = load_frag_b16(&Ks[(kt * 16) * 72], 72, ks * 32);
        a = wmma_bf16(qf[ks], bfr, a);
      }
      sc[kt] = a;
    }

    // scale + mask
    float sv[4][8];
    #pragma unroll
    for (int kt = 0; kt < 4; ++kt) {
      #pragma unroll
      for (int r = 0; r < 8; ++r) {
        float v = sc[kt][r] * 0.125f;  // 1/sqrt(64)
        int key = kv + kt * 16 + n0;
        int row = row_base + r + 8 * half;
        if (causal && key > row) v = -INFINITY;
        if (padmask[(size_t)b * S + key]) v = -INFINITY;
        sv[kt][r] = v;
      }
    }

    // online softmax (row reductions within 16-lane half: xor 1,2,4,8)
    bf16_t* Pw = &Ps[wave * 16 * 72];
    #pragma unroll
    for (int r = 0; r < 8; ++r) {
      float m = fmaxf(fmaxf(sv[0][r], sv[1][r]), fmaxf(sv[2][r], sv[3][r]));
      #pragma unroll
      for (int off = 1; off <= 8; off <<= 1) m = fmaxf(m, __shfl_xor(m, off, 32));
      float mn = fmaxf(m_i[r], m);
      float corr = __expf(m_i[r] - mn);
      m_i[r] = mn;
      float rs = 0.0f;
      #pragma unroll
      for (int kt = 0; kt < 4; ++kt) {
        float p = __expf(sv[kt][r] - mn);
        rs += p;
        Pw[(r + 8 * half) * 72 + kt * 16 + n0] = (bf16_t)p;
      }
      #pragma unroll
      for (int off = 1; off <= 8; off <<= 1) rs += __shfl_xor(rs, off, 32);
      l_i[r] = l_i[r] * corr + rs;
      #pragma unroll
      for (int dt = 0; dt < 4; ++dt) oacc[dt][r] *= corr;
    }

    // O += P V   (P re-fragmented from wave-local LDS; LDS in-order per wave)
    #pragma unroll
    for (int ks = 0; ks < 2; ++ks) {
      v16bf pf = load_frag_b16(Pw, 72, ks * 32);
      #pragma unroll
      for (int dt = 0; dt < 4; ++dt) {
        v16bf vf = load_frag_b16(&Vt[(dt * 16) * 72], 72, ks * 32);
        oacc[dt] = wmma_bf16(pf, vf, oacc[dt]);
      }
    }
  }

  // epilogue: normalize, write [B,S,H*64] bf16
  #pragma unroll
  for (int dt = 0; dt < 4; ++dt) {
    #pragma unroll
    for (int r = 0; r < 8; ++r) {
      int row = row_base + r + 8 * half;
      int d = dt * 16 + n0;
      float o = oacc[dt][r] / l_i[r];
      attn[((size_t)(b * S + row)) * 1024 + h * 64 + d] = (bf16_t)o;
    }
  }
}

// ---------------------------------------------------------------------------
// Kernel 4: output projection, fp32 result.
// ---------------------------------------------------------------------------
__global__ __launch_bounds__(256) void gemm_out_kernel(
    const bf16_t* __restrict__ A, const bf16_t* __restrict__ W,
    const float* __restrict__ bias, float* __restrict__ out,
    int lda, int ldb, int Ksz) {
  __shared__ bf16_t Xs[2 * TILE_ELEMS];
  __shared__ bf16_t Wt[2 * TILE_ELEMS];
  v8f acc[4][2] = {};
  const int rowBase = blockIdx.y * 128;
  const int colBase = blockIdx.x * 128;
  const int rows = (int)gridDim.y * 128;
  gemm_mainloop(A, W, lda, ldb, Ksz, rows, rowBase, colBase, Xs, Wt, acc);

  const int wave = threadIdx.x >> 5, lane = threadIdx.x & 31;
  const int wm = wave >> 2, wn = wave & 3;
  const int half = lane >> 4, n0 = lane & 15;
  #pragma unroll
  for (int mt = 0; mt < 4; ++mt) {
    #pragma unroll
    for (int nt = 0; nt < 2; ++nt) {
      int col = colBase + wn * 32 + nt * 16 + n0;
      float bv = bias[col];
      #pragma unroll
      for (int r = 0; r < 8; ++r) {
        int row = rowBase + wm * 64 + mt * 16 + r + 8 * half;
        out[(size_t)row * ldb + col] = acc[mt][nt][r] + bv;
      }
    }
  }
}

// ---------------------------------------------------------------------------
// Host launcher
// ---------------------------------------------------------------------------
extern "C" void kernel_launch(void* const* d_in, const int* in_sizes, int n_in,
                              void* d_out, int out_size, void* d_ws, size_t ws_size,
                              hipStream_t stream) {
  const float*         qkv  = (const float*)d_in[0];
  const unsigned char* pad  = (const unsigned char*)d_in[1];
  const float*         Wqkv = (const float*)d_in[2];
  const float*         bqkv = (const float*)d_in[3];
  const float*         Wout = (const float*)d_in[4];
  const float*         bout = (const float*)d_in[5];
  const int*           isc  = (const int*)d_in[6];
  float*               out  = (float*)d_out;

  const int S = 2048, D = 1024, H = 16;
  const int BS = in_sizes[0] / D;         // B * S
  const int B  = BS / S;

  uintptr_t base = (uintptr_t)d_ws;
  size_t off = 0;
  auto take = [&](size_t elems) {
    bf16_t* p = (bf16_t*)(base + off);
    off += ((elems * sizeof(bf16_t)) + 255) & ~(size_t)255;
    return p;
  };
  bf16_t* Xb     = take((size_t)BS * D);
  bf16_t* Wqkvb  = take((size_t)D * 3 * D);
  bf16_t* Woutb  = take((size_t)D * D);
  bf16_t* Qb     = take((size_t)BS * D);
  bf16_t* Kb     = take((size_t)BS * D);
  bf16_t* Vb     = take((size_t)BS * D);
  bf16_t* Attnb  = take((size_t)BS * D);
  (void)ws_size; (void)n_in; (void)out_size;

  {
    size_t n0 = (size_t)BS * D;
    cvt_f32_bf16<<<(unsigned)((n0 + 255) / 256), 256, 0, stream>>>(qkv, Xb, n0);
    size_t n1 = (size_t)D * 3 * D;
    cvt_f32_bf16<<<(unsigned)((n1 + 255) / 256), 256, 0, stream>>>(Wqkv, Wqkvb, n1);
    size_t n2 = (size_t)D * D;
    cvt_f32_bf16<<<(unsigned)((n2 + 255) / 256), 256, 0, stream>>>(Wout, Woutb, n2);
  }

  gemm_qkv_kernel<<<dim3(3 * D / 128, BS / 128), 256, 0, stream>>>(
      Xb, Wqkvb, bqkv, Qb, Kb, Vb, S, D, 3 * D);

  {
    size_t n = 2ull * B * H * S * 32;
    rope_kernel<<<(unsigned)((n + 255) / 256), 256, 0, stream>>>(Qb, Kb, B * H, S);
  }

  flash_attn_kernel<<<dim3(S / 128, B * H), 256, 0, stream>>>(
      Qb, Kb, Vb, pad, isc, Attnb, S);

  gemm_out_kernel<<<dim3(D / 128, BS / 128), 256, 0, stream>>>(
      Attnb, Woutb, bout, out, D, D, D);
}
